// TransLayer_23579370455061
// MI455X (gfx1250) — matrix operand order repaired and checked by reading, not compiled, not run on hardware
//
#include <hip/hip_runtime.h>

// ---------------- problem constants ----------------
#define BB 4
#define NN 8192
#define DIMV 512
#define HH 8
#define DD 64
#define MM 256
#define BH (BB*HH)
#define KERN 33

typedef __bf16 bf16;
typedef __attribute__((ext_vector_type(16))) __bf16 v16bf;
typedef __attribute__((ext_vector_type(8)))  float  v8f;
typedef __attribute__((ext_vector_type(2)))  float  v2f;

// ---------------- WMMA wrappers ----------------
__device__ __forceinline__ v8f wmma_bf16(v16bf a, v16bf b, v8f c) {
  return __builtin_amdgcn_wmma_f32_16x16x32_bf16(false, a, false, b, (short)0, c, false, false);
}
__device__ __forceinline__ v8f wmma_f32(v2f a, v2f b, v8f c) {
  return __builtin_amdgcn_wmma_f32_16x16x4_f32(false, a, false, b, (short)0, c, false, false);
}
__device__ __forceinline__ v8f zero8() { v8f z = {0.f,0.f,0.f,0.f,0.f,0.f,0.f,0.f}; return z; }

// ---------------- fragment loads (wave32; bf16 A/B use contiguous k = 16*hi + e) ----------------
// A 16x32 bf16, row-major source with leading dim ldk.
// lanes 0-15: row=lane; K=0..15
// lanes16-31: row=lane-16; K=16..31
__device__ __forceinline__ v16bf ld_a_mk(const bf16* base, int ldk) {
  int lane = threadIdx.x & 31, row = lane & 15, hi = lane >> 4;
  v16bf a;
#pragma unroll
  for (int e = 0; e < 16; ++e) {
    a[e] = base[(size_t)row * ldk + hi * 16 + e];
  }
  return a;
}
// B 32x16 bf16 from K-major (KxN row-major, leading dim ldn):
// lane<16: col=lane, K=0..15 ; lane>=16: col=lane-16, K=16..31
__device__ __forceinline__ v16bf ld_b_kn(const bf16* base, int ldn) {
  int lane = threadIdx.x & 31, col = lane & 15, hi = (lane >> 4) * 16;
  v16bf b;
#pragma unroll
  for (int e = 0; e < 16; ++e) b[e] = base[(size_t)(hi + e) * ldn + col];
  return b;
}
// B 32x16 from N-major source (NxK row-major, i.e. computing A @ Src^T)
__device__ __forceinline__ v16bf ld_b_nk(const bf16* base, int ldk) {
  int lane = threadIdx.x & 31, col = lane & 15, hi = (lane >> 4) * 16;
  v16bf b;
#pragma unroll
  for (int e = 0; e < 16; ++e) b[e] = base[(size_t)col * ldk + hi + e];
  return b;
}
// f32 16x16x4 frags: A 16x4 (a[j]: K = hi*2+j), B 4x16 (b[j]: K = hi*2+j)
__device__ __forceinline__ v2f ld_a_f32(const float* base, int ldk, int k) {
  int lane = threadIdx.x & 31, row = lane & 15, kk = k + ((lane >> 4) << 1);
  v2f a; a[0] = base[(size_t)row * ldk + kk]; a[1] = base[(size_t)row * ldk + kk + 1];
  return a;
}
__device__ __forceinline__ v2f ld_b_f32(const float* base, int ldn, int k) {
  int lane = threadIdx.x & 31, col = lane & 15, kk = k + ((lane >> 4) << 1);
  v2f b; b[0] = base[(size_t)kk * ldn + col]; b[1] = base[(size_t)(kk + 1) * ldn + col];
  return b;
}
// C/D 16x16 f32: lane<16 rows 0..7, lane>=16 rows 8..15 (per VGPR r)
__device__ __forceinline__ void st_c(float* C, int ldc, const v8f& c) {
  int lane = threadIdx.x & 31, col = lane & 15, rb = (lane >> 4) * 8;
#pragma unroll
  for (int r = 0; r < 8; ++r) C[(size_t)(rb + r) * ldc + col] = c[r];
}

__device__ __forceinline__ float wave_max(float v) {
#pragma unroll
  for (int m = 16; m; m >>= 1) v = fmaxf(v, __shfl_xor(v, m, 32));
  return v;
}
__device__ __forceinline__ float wave_sum(float v) {
#pragma unroll
  for (int m = 16; m; m >>= 1) v += __shfl_xor(v, m, 32);
  return v;
}

// ---------------- kernels ----------------

// LayerNorm: x (B*N,512) -> bf16 nx
__global__ __launch_bounds__(128) void k_ln(const float* __restrict__ x,
                                            const float* __restrict__ g,
                                            const float* __restrict__ bta,
                                            bf16* __restrict__ nxb) {
  int row = blockIdx.x, t = threadIdx.x;
  const float* xr = x + (size_t)row * DIMV;
  float4 v = ((const float4*)xr)[t];
  float s = v.x + v.y + v.z + v.w;
  float q = v.x * v.x + v.y * v.y + v.z * v.z + v.w * v.w;
  __shared__ float sb[8];
  s = wave_sum(s); q = wave_sum(q);
  int w = t >> 5, l = t & 31;
  if (l == 0) { sb[w] = s; sb[4 + w] = q; }
  __syncthreads();
  s = sb[0] + sb[1] + sb[2] + sb[3];
  q = sb[4] + sb[5] + sb[6] + sb[7];
  float mu = s * (1.f / DIMV);
  float rs = rsqrtf(q * (1.f / DIMV) - mu * mu + 1e-5f);
  float4 gg = ((const float4*)g)[t];
  float4 bb = ((const float4*)bta)[t];
  bf16* o = nxb + (size_t)row * DIMV + t * 4;
  o[0] = (bf16)((v.x - mu) * rs * gg.x + bb.x);
  o[1] = (bf16)((v.y - mu) * rs * gg.y + bb.y);
  o[2] = (bf16)((v.z - mu) * rs * gg.z + bb.z);
  o[3] = (bf16)((v.w - mu) * rs * gg.w + bb.w);
}

__global__ __launch_bounds__(256) void k_cvt_bf16(const float* __restrict__ s,
                                                  bf16* __restrict__ d, int n) {
  int i = blockIdx.x * 256 + threadIdx.x;
  if (i < n) d[i] = (bf16)s[i];
}

__global__ __launch_bounds__(256) void k_zero(float* p, int n) {
  int i = blockIdx.x * 256 + threadIdx.x;
  if (i < n) p[i] = 0.f;
}

// QKV GEMM: nx(32768x512) @ wqkv(512x1536) -> scatter q(bf16,*0.125), k(bf16), v(f32) head-major
__global__ __launch_bounds__(256) void k_gemm_qkv(const bf16* __restrict__ nxb,
                                                  const bf16* __restrict__ wq,
                                                  bf16* __restrict__ qb, bf16* __restrict__ kb,
                                                  float* __restrict__ vf) {
  __shared__ bf16 As[128 * 32];
  __shared__ bf16 Bs[32 * 64];
  int tid = threadIdx.x, wid = tid >> 5, lane = tid & 31;
  int row0 = blockIdx.y * 128, col0 = blockIdx.x * 64;
  int wm = wid >> 1, wn = wid & 1;
  v8f acc[2][2];
#pragma unroll
  for (int i = 0; i < 2; i++)
#pragma unroll
    for (int j = 0; j < 2; j++) acc[i][j] = zero8();
  for (int k0 = 0; k0 < DIMV; k0 += 32) {
#pragma unroll
    for (int i = 0; i < 8; i++) {
      int idx = tid + i * 256, r = idx >> 4, cc = idx & 15;
      ((unsigned int*)As)[idx] = ((const unsigned int*)(nxb + (size_t)(row0 + r) * DIMV))[(k0 >> 1) + cc];
    }
#pragma unroll
    for (int i = 0; i < 4; i++) {
      int idx = tid + i * 256, r = idx >> 5, cc = idx & 31;
      ((unsigned int*)Bs)[idx] = ((const unsigned int*)(wq + (size_t)(k0 + r) * (3 * DIMV)))[(col0 >> 1) + cc];
    }
    // prefetch next K panel toward the WGP while this tile computes (global_prefetch_b8)
    if (k0 + 32 < DIMV) {
      __builtin_prefetch(nxb + (size_t)(row0 + (tid & 127)) * DIMV + k0 + 32, 0, 0);
      __builtin_prefetch(wq + (size_t)(k0 + 32 + (tid >> 3)) * (3 * DIMV) + col0 + ((tid & 7) << 3), 0, 0);
    }
    __syncthreads();
    v16bf bf[2];
#pragma unroll
    for (int j = 0; j < 2; j++) bf[j] = ld_b_kn(Bs + wn * 32 + j * 16, 64);
#pragma unroll
    for (int i = 0; i < 2; i++) {
      v16bf a = ld_a_mk(As + (size_t)(wm * 32 + i * 16) * 32, 32);
#pragma unroll
      for (int j = 0; j < 2; j++) acc[i][j] = wmma_bf16(a, bf[j], acc[i][j]);
    }
    __syncthreads();
  }
  int rb = (lane >> 4) * 8, col = lane & 15;
#pragma unroll
  for (int i = 0; i < 2; i++)
#pragma unroll
    for (int j = 0; j < 2; j++)
#pragma unroll
      for (int r = 0; r < 8; r++) {
        int grow = row0 + wm * 32 + i * 16 + rb + r;
        int gcol = col0 + wn * 32 + j * 16 + col;
        float val = acc[i][j][r];
        int which = gcol >> 9, ci = gcol & 511;
        int h = ci >> 6, dd = ci & 63;
        int b = grow >> 13, n = grow & (NN - 1);
        size_t o = (((size_t)(b * HH + h)) * NN + n) * DD + dd;
        if (which == 0) qb[o] = (bf16)(val * 0.125f);
        else if (which == 1) kb[o] = (bf16)val;
        else vf[o] = val;
      }
}

// landmark means over groups of 32 tokens
__global__ __launch_bounds__(256) void k_landmarks(const bf16* __restrict__ qb, const bf16* __restrict__ kb,
                                                   float* __restrict__ ql, float* __restrict__ kl,
                                                   bf16* __restrict__ qlb, bf16* __restrict__ klb) {
  int id = blockIdx.x * 256 + threadIdx.x;   // BH*M*D
  int dd = id & 63, m = (id >> 6) & 255, bh = id >> 14;
  size_t base = ((size_t)bh * NN + m * 32) * DD + dd;
  float sq = 0.f, sk = 0.f;
#pragma unroll
  for (int j = 0; j < 32; j++) { sq += (float)qb[base + (size_t)j * DD]; sk += (float)kb[base + (size_t)j * DD]; }
  sq *= (1.f / 32.f); sk *= (1.f / 32.f);
  ql[id] = sq; kl[id] = sk; qlb[id] = (bf16)sq; klb[id] = (bf16)sk;
}

// attn2 = softmax(q_l @ k_l^T), per bh; 32-row blocks
__global__ __launch_bounds__(256) void k_attn2(const bf16* __restrict__ qlb, const bf16* __restrict__ klb,
                                               float* __restrict__ a2) {
  int bh = blockIdx.y, r0 = blockIdx.x * 32;
  __shared__ float Sf[32 * 256];
  int tid = threadIdx.x, wid = tid >> 5, lane = tid & 31;
  const bf16* Ab = qlb + ((size_t)bh * MM + r0) * DD;
  const bf16* Kl = klb + (size_t)bh * MM * DD;
#pragma unroll
  for (int i = 0; i < 4; i++) {
    int tt = wid + i * 8, rt = tt >> 4, ct = tt & 15;
    v8f acc = zero8();
#pragma unroll
    for (int kk = 0; kk < 2; kk++) {
      v16bf a = ld_a_mk(Ab + (size_t)(rt * 16) * DD + kk * 32, DD);
      v16bf b = ld_b_nk(Kl + (size_t)(ct * 16) * DD + kk * 32, DD);
      acc = wmma_bf16(a, b, acc);
    }
    st_c(&Sf[(size_t)(rt * 16) * 256 + ct * 16], 256, acc);
  }
  __syncthreads();
#pragma unroll
  for (int rr = 0; rr < 4; rr++) {
    int row = wid * 4 + rr;
    float vals[8], mx = -1e30f;
#pragma unroll
    for (int j = 0; j < 8; j++) { vals[j] = Sf[row * 256 + lane + j * 32]; mx = fmaxf(mx, vals[j]); }
    mx = wave_max(mx);
    float s = 0.f;
#pragma unroll
    for (int j = 0; j < 8; j++) { vals[j] = __expf(vals[j] - mx); s += vals[j]; }
    s = wave_sum(s);
    float inv = 1.f / s;
#pragma unroll
    for (int j = 0; j < 8; j++) a2[((size_t)bh * MM + r0 + row) * MM + lane + j * 32] = vals[j] * inv;
  }
}

// global abs row-sum / col-sum max for pinv init scale (nonneg floats -> int atomicMax trick)
__global__ __launch_bounds__(256) void k_scale(const float* __restrict__ a2, float* __restrict__ scal) {
  int id = blockIdx.x * 256 + threadIdx.x;   // BH*2*256
  int bh = id >> 9, rem = id & 511, which = rem >> 8, idx = rem & 255;
  const float* Ab = a2 + (size_t)bh * (MM * MM);
  float s = 0.f;
  if (which == 0) { for (int j = 0; j < MM; j++) s += fabsf(Ab[idx * MM + j]); }
  else            { for (int i = 0; i < MM; i++) s += fabsf(Ab[i * MM + idx]); }
  atomicMax((int*)(scal + which), __float_as_int(s));
}

__global__ __launch_bounds__(256) void k_pinv_init(const float* __restrict__ a2, const float* __restrict__ scal,
                                                   float* __restrict__ z0) {
  int id = blockIdx.x * 256 + threadIdx.x;  // BH*256*256
  float inv = 1.f / (scal[0] * scal[1]);
  int bh = id >> 16, r = (id >> 8) & 255, c = id & 255;
  z0[id] = a2[((size_t)bh << 16) + c * 256 + r] * inv;
}

__global__ __launch_bounds__(256) void k_sub7I(const float* __restrict__ in, float* __restrict__ out) {
  int id = blockIdx.x * 256 + threadIdx.x;
  int r = (id >> 8) & 255, c = id & 255;
  out[id] = ((r == c) ? 7.f : 0.f) - in[id];
}

// batched 256x256x256 f32 GEMM: Out = alpha*(A@B) + diag*I, via v_wmma_f32_16x16x4_f32
__global__ __launch_bounds__(256) void k_gemm256(const float* __restrict__ A, const float* __restrict__ Bm,
                                                 float* __restrict__ Out, float alpha, float diag) {
  int bh = blockIdx.y, r0 = blockIdx.x * 64;
  const float* Ab = A + ((size_t)bh << 16);
  const float* Bb = Bm + ((size_t)bh << 16);
  int tid = threadIdx.x, wid = tid >> 5, lane = tid & 31;
  v8f acc[8];
#pragma unroll
  for (int i = 0; i < 8; i++) acc[i] = zero8();
  for (int k = 0; k < 256; k += 4) {
    if (k + 16 < 256) {
      __builtin_prefetch(Bb + (size_t)(k + 16 + (tid >> 6)) * 256 + ((tid & 63) << 2), 0, 0);
    }
    v2f a4[4], b2[2];
#pragma unroll
    for (int rt = 0; rt < 4; rt++) a4[rt] = ld_a_f32(Ab + (size_t)(r0 + rt * 16) * 256, 256, k);
#pragma unroll
    for (int j = 0; j < 2; j++) b2[j] = ld_b_f32(Bb + (wid + j * 8) * 16, 256, k);
#pragma unroll
    for (int i = 0; i < 8; i++) acc[i] = wmma_f32(a4[i >> 1], b2[i & 1], acc[i]);
  }
  int rb = (lane >> 4) * 8, col = lane & 15;
#pragma unroll
  for (int i = 0; i < 8; i++) {
    int rt = i >> 1, ctb = (wid + (i & 1) * 8) * 16;
#pragma unroll
    for (int r = 0; r < 8; r++) {
      int gr = r0 + rt * 16 + rb + r, gc = ctb + col;
      Out[((size_t)bh << 16) + (size_t)gr * 256 + gc] = alpha * acc[i][r] + ((gr == gc) ? diag : 0.f);
    }
  }
}

// W = softmax(q_l @ k^T, axis=n) @ v : two-pass online softmax, WMMA both passes
__global__ __launch_bounds__(256) void k_flash(const bf16* __restrict__ qlb, const bf16* __restrict__ kb,
                                               const float* __restrict__ vf,
                                               float* __restrict__ W, bf16* __restrict__ Wb) {
  int bh = blockIdx.y, r0 = blockIdx.x * 64;
  __shared__ float Sf[64 * 64];
  __shared__ bf16 Pb[64 * 64];
  __shared__ bf16 Vb[64 * 64];
  __shared__ float mrow[64], srow[64];
  int tid = threadIdx.x, wid = tid >> 5, lane = tid & 31;
  if (tid < 64) { mrow[tid] = -1e30f; srow[tid] = 0.f; }
  __syncthreads();
  const bf16* Ab = qlb + ((size_t)bh * MM + r0) * DD;
  const bf16* Kb = kb + (size_t)bh * NN * DD;
  // pass 1: stats
  for (int c0 = 0; c0 < NN; c0 += 64) {
    // prefetch next k panel (64 rows x 64 bf16 = 64 lines of 128B)
    if (c0 + 64 < NN)
      __builtin_prefetch(Kb + (size_t)(c0 + 64 + (tid >> 2)) * DD + ((tid & 3) << 4), 0, 0);
#pragma unroll
    for (int i = 0; i < 2; i++) {
      int tt = wid + i * 8, rt = tt >> 2, ct = tt & 3;
      v8f acc = zero8();
#pragma unroll
      for (int kk = 0; kk < 2; kk++) {
        v16bf a = ld_a_mk(Ab + (size_t)(rt * 16) * DD + kk * 32, DD);
        v16bf b = ld_b_nk(Kb + (size_t)(c0 + ct * 16) * DD + kk * 32, DD);
        acc = wmma_bf16(a, b, acc);
      }
      st_c(&Sf[(size_t)(rt * 16) * 64 + ct * 16], 64, acc);
    }
    __syncthreads();
#pragma unroll
    for (int rr = 0; rr < 8; rr++) {
      int row = wid * 8 + rr;
      float v0 = Sf[row * 64 + lane], v1 = Sf[row * 64 + 32 + lane];
      float cm = wave_max(fmaxf(v0, v1));
      float mo = mrow[row], mn = fmaxf(mo, cm);
      float es = wave_sum(__expf(v0 - mn) + __expf(v1 - mn));
      if (lane == 0) { srow[row] = srow[row] * __expf(mo - mn) + es; mrow[row] = mn; }
    }
    __syncthreads();
  }
  // pass 2: accumulate exp(S-m) @ v with WMMA C-chains
  v8f acc2[2]; acc2[0] = zero8(); acc2[1] = zero8();
  for (int c0 = 0; c0 < NN; c0 += 64) {
    if (c0 + 64 < NN) {
      __builtin_prefetch(Kb + (size_t)(c0 + 64 + (tid >> 2)) * DD + ((tid & 3) << 4), 0, 0);
      __builtin_prefetch(vf + ((size_t)bh * NN + c0 + 64 + (tid >> 2)) * DD + ((tid & 3) << 4), 0, 0);
    }
#pragma unroll
    for (int i = 0; i < 2; i++) {
      int tt = wid + i * 8, rt = tt >> 2, ct = tt & 3;
      v8f acc = zero8();
#pragma unroll
      for (int kk = 0; kk < 2; kk++) {
        v16bf a = ld_a_mk(Ab + (size_t)(rt * 16) * DD + kk * 32, DD);
        v16bf b = ld_b_nk(Kb + (size_t)(c0 + ct * 16) * DD + kk * 32, DD);
        acc = wmma_bf16(a, b, acc);
      }
      st_c(&Sf[(size_t)(rt * 16) * 64 + ct * 16], 64, acc);
    }
    __syncthreads();
#pragma unroll
    for (int i = 0; i < 16; i++) {
      int idx = tid + i * 256, row = idx >> 6, cc = idx & 63;
      Pb[idx] = (bf16)__expf(Sf[idx] - mrow[row]);
      Vb[idx] = (bf16)vf[((size_t)bh * NN + c0 + row) * DD + cc];
    }
    __syncthreads();
#pragma unroll
    for (int i = 0; i < 2; i++) {
      int tt = wid + i * 8, rt = tt >> 2, ct = tt & 3;
#pragma unroll
      for (int kk = 0; kk < 2; kk++) {
        v16bf a = ld_a_mk(Pb + (size_t)(rt * 16) * 64 + kk * 32, 64);
        v16bf b = ld_b_kn(Vb + (size_t)(kk * 32) * 64 + ct * 16, 64);
        acc2[i] = wmma_bf16(a, b, acc2[i]);
      }
    }
    __syncthreads();
  }
  int rb = (lane >> 4) * 8, col = lane & 15;
#pragma unroll
  for (int i = 0; i < 2; i++) {
    int tt = wid + i * 8, rt = tt >> 2, ct = tt & 3;
#pragma unroll
    for (int r = 0; r < 8; r++) {
      int lr = rt * 16 + rb + r;
      float val = acc2[i][r] / srow[lr];
      size_t o = ((size_t)bh * MM + r0 + lr) * DD + ct * 16 + col;
      W[o] = val; Wb[o] = (bf16)val;
    }
  }
}

// fused: O = softmax(q @ k_l^T) @ Z @ W  -> aout merged (b,n,512)
__global__ __launch_bounds__(256) void k_attn_main(const bf16* __restrict__ qb, const bf16* __restrict__ klb,
                                                   const bf16* __restrict__ zbf, const bf16* __restrict__ wbf,
                                                   float* __restrict__ aout) {
  int bh = blockIdx.y, n0 = blockIdx.x * 32;
  int b = bh >> 3, h = bh & 7;
  __shared__ float Sf[32 * 256];
  __shared__ bf16 Pb[32 * 256];
  bf16* P2b = (bf16*)Sf;   // alias: Sf dead after softmax
  int tid = threadIdx.x, wid = tid >> 5, lane = tid & 31;
  const bf16* Ab = qb + ((size_t)bh * NN + n0) * DD;
  const bf16* Kl = klb + (size_t)bh * MM * DD;
  // keep the L2-resident Z panel warm for stage 2 (global_prefetch_b8)
  __builtin_prefetch(zbf + ((size_t)bh << 16) + (size_t)tid * 256, 0, 0);
  // stage 1: logits
#pragma unroll
  for (int i = 0; i < 4; i++) {
    int tt = wid + i * 8, rt = tt >> 4, ct = tt & 15;
    v8f acc = zero8();
#pragma unroll
    for (int kk = 0; kk < 2; kk++) {
      v16bf a = ld_a_mk(Ab + (size_t)(rt * 16) * DD + kk * 32, DD);
      v16bf b = ld_b_nk(Kl + (size_t)(ct * 16) * DD + kk * 32, DD);
      acc = wmma_bf16(a, b, acc);
    }
    st_c(&Sf[(size_t)(rt * 16) * 256 + ct * 16], 256, acc);
  }
  __syncthreads();
  // softmax -> Pb (normalized, bf16)
#pragma unroll
  for (int rr = 0; rr < 4; rr++) {
    int row = wid * 4 + rr;
    float vals[8], mx = -1e30f;
#pragma unroll
    for (int j = 0; j < 8; j++) { vals[j] = Sf[row * 256 + lane + j * 32]; mx = fmaxf(mx, vals[j]); }
    mx = wave_max(mx);
    float s = 0.f;
#pragma unroll
    for (int j = 0; j < 8; j++) { vals[j] = __expf(vals[j] - mx); s += vals[j]; }
    s = wave_sum(s);
    float inv = 1.f / s;
#pragma unroll
    for (int j = 0; j < 8; j++) Pb[row * 256 + lane + j * 32] = (bf16)(vals[j] * inv);
  }
  __syncthreads();
  // stage 2: P2 = Pb @ Z
#pragma unroll
  for (int i = 0; i < 4; i++) {
    int tt = wid + i * 8, rt = tt >> 4, ct = tt & 15;
    v8f acc = zero8();
#pragma unroll
    for (int kk = 0; kk < 8; kk++) {
      v16bf a = ld_a_mk(Pb + (size_t)(rt * 16) * 256 + kk * 32, 256);
      v16bf b = ld_b_kn(zbf + ((size_t)bh << 16) + (size_t)(kk * 32) * 256 + ct * 16, 256);
      acc = wmma_bf16(a, b, acc);
    }
    int rb = (lane >> 4) * 8, col = lane & 15;
#pragma unroll
    for (int r = 0; r < 8; r++) P2b[(rt * 16 + rb + r) * 256 + ct * 16 + col] = (bf16)acc[r];
  }
  __syncthreads();
  // stage 3: O = P2 @ W
  {
    int rt = wid >> 2, ct = wid & 3;
    v8f acc = zero8();
#pragma unroll
    for (int kk = 0; kk < 8; kk++) {
      v16bf a = ld_a_mk(P2b + (size_t)(rt * 16) * 256 + kk * 32, 256);
      v16bf b = ld_b_kn(wbf + (size_t)bh * (MM * DD) + (size_t)(kk * 32) * DD + ct * 16, DD);
      acc = wmma_bf16(a, b, acc);
    }
    int rb = (lane >> 4) * 8, col = lane & 15;
#pragma unroll
    for (int r = 0; r < 8; r++) {
      int n = n0 + rt * 16 + rb + r;
      aout[((size_t)b * NN + n) * DIMV + h * DD + ct * 16 + col] = acc[r];
    }
  }
}

// depthwise conv residual over sequence (kernel 33), added into aout
__global__ __launch_bounds__(256) void k_conv(const float* __restrict__ vf, const float* __restrict__ kern,
                                              float* __restrict__ aout) {
  int id = blockIdx.x * 256 + threadIdx.x;   // B*N*512
  int c = id & 511, n = (id >> 9) & (NN - 1), b = id >> 22;
  int h = c >> 6, dd = c & 63;
  const float* kr = kern + h * KERN;
  size_t vbase = ((size_t)(b * HH + h) * NN) * DD + dd;
  float s = 0.f;
#pragma unroll
  for (int t = 0; t < KERN; t++) {
    int nn = n + t - 16;
    if (nn >= 0 && nn < NN) s += kr[t] * vf[vbase + (size_t)nn * DD];
  }
  aout[id] += s;
}

// out projection: aout(32768x512) @ wout(512x512) + b_out + omega*x -> d_out
__global__ __launch_bounds__(256) void k_gemm_out(const float* __restrict__ aout, const bf16* __restrict__ wob,
                                                  const float* __restrict__ x, const float* __restrict__ bout,
                                                  const float* __restrict__ omega, float* __restrict__ out) {
  __shared__ bf16 As[128 * 32];
  __shared__ bf16 Bs[32 * 64];
  int tid = threadIdx.x, wid = tid >> 5, lane = tid & 31;
  int row0 = blockIdx.y * 128, col0 = blockIdx.x * 64;
  int wm = wid >> 1, wn = wid & 1;
  v8f acc[2][2];
#pragma unroll
  for (int i = 0; i < 2; i++)
#pragma unroll
    for (int j = 0; j < 2; j++) acc[i][j] = zero8();
  for (int k0 = 0; k0 < DIMV; k0 += 32) {
#pragma unroll
    for (int i = 0; i < 16; i++) {
      int idx = tid + i * 256, r = idx >> 5, cc = idx & 31;
      As[idx] = (bf16)aout[(size_t)(row0 + r) * DIMV + k0 + cc];
    }
#pragma unroll
    for (int i = 0; i < 4; i++) {
      int idx = tid + i * 256, r = idx >> 5, cc = idx & 31;
      ((unsigned int*)Bs)[idx] = ((const unsigned int*)(wob + (size_t)(k0 + r) * DIMV))[(col0 >> 1) + cc];
    }
    if (k0 + 32 < DIMV) {
      __builtin_prefetch(aout + (size_t)(row0 + (tid & 127)) * DIMV + k0 + 32, 0, 0);
      __builtin_prefetch(wob + (size_t)(k0 + 32 + (tid >> 3)) * DIMV + col0 + ((tid & 7) << 3), 0, 0);
    }
    __syncthreads();
    v16bf bf[2];
#pragma unroll
    for (int j = 0; j < 2; j++) bf[j] = ld_b_kn(Bs + wn * 32 + j * 16, 64);
#pragma unroll
    for (int i = 0; i < 2; i++) {
      v16bf a = ld_a_mk(As + (size_t)(wm * 32 + i * 16) * 32, 32);
#pragma unroll
      for (int j = 0; j < 2; j++) acc[i][j] = wmma_bf16(a, bf[j], acc[i][j]);
    }
    __syncthreads();
  }
  float om = omega[0];
  int rb = (lane >> 4) * 8, col = lane & 15;
#pragma unroll
  for (int i = 0; i < 2; i++)
#pragma unroll
    for (int j = 0; j < 2; j++)
#pragma unroll
      for (int r = 0; r < 8; r++) {
        int grow = row0 + wm * 32 + i * 16 + rb + r;
        int gcol = col0 + wn * 32 + j * 16 + col;
        size_t o = (size_t)grow * DIMV + gcol;
        out[o] = acc[i][j][r] + bout[gcol] + om * x[o];
      }
}

// ---------------- host launcher ----------------
extern "C" void kernel_launch(void* const* d_in, const int* in_sizes, int n_in,
                              void* d_out, int out_size, void* d_ws, size_t ws_size,
                              hipStream_t stream) {
  (void)in_sizes; (void)n_in; (void)out_size; (void)ws_size;
  const float* x     = (const float*)d_in[0];
  const float* ln_g  = (const float*)d_in[1];
  const float* ln_b  = (const float*)d_in[2];
  const float* w_qkv = (const float*)d_in[3];
  const float* w_out = (const float*)d_in[4];
  const float* b_out = (const float*)d_in[5];
  const float* rkern = (const float*)d_in[6];
  const float* omega = (const float*)d_in[7];
  float* out = (float*)d_out;

  char* w = (char*)d_ws;
  size_t off = 0;
  auto alloc = [&](size_t bytes) { char* p = w + off; off += (bytes + 255) & ~(size_t)255; return p; };
  bf16*  NXB  = (bf16*) alloc((size_t)BB * NN * DIMV * 2);
  bf16*  QB   = (bf16*) alloc((size_t)BB * NN * DIMV * 2);
  bf16*  KB   = (bf16*) alloc((size_t)BB * NN * DIMV * 2);
  float* VF   = (float*)alloc((size_t)BB * NN * DIMV * 4);
  bf16*  WQB  = (bf16*) alloc((size_t)DIMV * 3 * DIMV * 2);
  bf16*  WOB  = (bf16*) alloc((size_t)DIMV * DIMV * 2);
  float* QL   = (float*)alloc((size_t)BH * MM * DD * 4);
  float* KL   = (float*)alloc((size_t)BH * MM * DD * 4);
  bf16*  QLB  = (bf16*) alloc((size_t)BH * MM * DD * 2);
  bf16*  KLB  = (bf16*) alloc((size_t)BH * MM * DD * 2);
  float* A2   = (float*)alloc((size_t)BH * MM * MM * 4);
  float* AZ   = (float*)alloc((size_t)BH * MM * MM * 4);
  float* T1   = (float*)alloc((size_t)BH * MM * MM * 4);
  float* T2   = (float*)alloc((size_t)BH * MM * MM * 4);
  float* Z0   = (float*)alloc((size_t)BH * MM * MM * 4);
  float* Z1   = (float*)alloc((size_t)BH * MM * MM * 4);
  bf16*  ZBF  = (bf16*) alloc((size_t)BH * MM * MM * 2);
  float* WT   = (float*)alloc((size_t)BH * MM * DD * 4);
  bf16*  WTB  = (bf16*) alloc((size_t)BH * MM * DD * 2);
  float* AOUT = (float*)alloc((size_t)BB * NN * DIMV * 4);
  float* SCAL = (float*)alloc(256);
  (void)QL; (void)KL;

  // 1. weights to bf16, layernorm
  k_cvt_bf16<<<(DIMV * 3 * DIMV + 255) / 256, 256, 0, stream>>>(w_qkv, WQB, DIMV * 3 * DIMV);
  k_cvt_bf16<<<(DIMV * DIMV + 255) / 256, 256, 0, stream>>>(w_out, WOB, DIMV * DIMV);
  k_ln<<<BB * NN, 128, 0, stream>>>(x, ln_g, ln_b, NXB);
  // 2. QKV projection
  k_gemm_qkv<<<dim3(3 * DIMV / 64, BB * NN / 128), 256, 0, stream>>>(NXB, WQB, QB, KB, VF);
  // 3. landmarks + attn2 + softmax
  k_landmarks<<<(BH * MM * DD) / 256, 256, 0, stream>>>(QB, KB, QL, KL, QLB, KLB);
  k_attn2<<<dim3(MM / 32, BH), 256, 0, stream>>>(QLB, KLB, A2);
  // 4. Moore-Penrose pinv (f32 WMMA)
  k_zero<<<1, 256, 0, stream>>>(SCAL, 2);
  k_scale<<<(BH * 2 * MM) / 256, 256, 0, stream>>>(A2, SCAL);
  k_pinv_init<<<(BH * MM * MM) / 256, 256, 0, stream>>>(A2, SCAL, Z0);
  float* zin = Z0; float* zout = Z1;
  int elems = BH * MM * MM;
  for (int it = 0; it < 6; ++it) {
    k_gemm256<<<dim3(MM / 64, BH), 256, 0, stream>>>(A2, zin, AZ, 1.f, 0.f);       // az = a@z
    k_sub7I<<<elems / 256, 256, 0, stream>>>(AZ, T1);                               // t1 = 7I - az
    k_gemm256<<<dim3(MM / 64, BH), 256, 0, stream>>>(AZ, T1, T2, -1.f, 15.f);      // t2 = 15I - az@t1
    k_gemm256<<<dim3(MM / 64, BH), 256, 0, stream>>>(AZ, T2, T1, -1.f, 13.f);      // t3 = 13I - az@t2
    k_gemm256<<<dim3(MM / 64, BH), 256, 0, stream>>>(zin, T1, zout, 0.25f, 0.f);   // z = 0.25 z@t3
    float* t = zin; zin = zout; zout = t;
  }
  k_cvt_bf16<<<elems / 256, 256, 0, stream>>>(zin, ZBF, elems);
  // 5. W = softmax(q_l k^T) @ v  (flash, two-pass)
  k_flash<<<dim3(MM / 64, BH), 256, 0, stream>>>(QLB, KB, VF, WT, WTB);
  // 6. fused attn1 @ Z @ W
  k_attn_main<<<dim3(NN / 32, BH), 256, 0, stream>>>(QB, KLB, ZBF, WTB, AOUT);
  // 7. conv residual
  k_conv<<<(BB * NN * DIMV) / 256, 256, 0, stream>>>(VF, rkern, AOUT);
  // 8. output projection + bias + omega*x
  k_gemm_out<<<dim3(DIMV / 64, BB * NN / 128), 256, 0, stream>>>(AOUT, WOB, x, b_out, omega, out);
}
